// GNNModel_50620484550702
// MI455X (gfx1250) — matrix operand order repaired and checked by reading, not compiled, not run on hardware
//
#include <hip/hip_runtime.h>

// ---------------------------------------------------------------------------
// GNN forward for MI455X (gfx1250, wave32).
// Per-edge linears hoisted out of the segment-sum (linearity) -> edges are a
// pure f32 gather + atomic-scatter (bandwidth bound, ~3 GB total @ 23.3 TB/s),
// node-level GEMMs run on the matrix pipes via v_wmma_f32_16x16x32_bf16 with
// LDS-staged bf16 A-tiles and pre-converted bf16 weights.
// ---------------------------------------------------------------------------

#define DD 128          // feature dim
#define GG 512          // number of graphs (fixed by reference)
#define OUTD 64
#define APAD 136        // padded LDS row stride (bf16 elems) to spread banks

typedef __attribute__((ext_vector_type(16))) __bf16 v16bf;
typedef __attribute__((ext_vector_type(8)))  __bf16 v8bf;
typedef __attribute__((ext_vector_type(8)))  float  v8f;

// ---------------------------------------------------------------------------
// h = emb[nodes]   (thread = node x 4 features)
__global__ __launch_bounds__(256) void k_gather(const int* __restrict__ nodes,
                                                const float* __restrict__ emb,
                                                float* __restrict__ H, int n) {
    int t = blockIdx.x * blockDim.x + threadIdx.x;
    if (t >= n * 32) return;
    int i = t >> 5, d = (t & 31) * 4;
    const float4 v = *(const float4*)(emb + (size_t)nodes[i] * DD + d);
    *(float4*)(H + (size_t)i * DD + d) = v;
}

// deg[n] = #(dst==n) + #(src==n)
__global__ __launch_bounds__(256) void k_degree(const int* __restrict__ src,
                                                const int* __restrict__ dst,
                                                int* __restrict__ deg, int E) {
    int e = blockIdx.x * blockDim.x + threadIdx.x;
    if (e >= E) return;
    atomicAdd(&deg[dst[e]], 1);
    atomicAdd(&deg[src[e]], 1);
}

__global__ __launch_bounds__(256) void k_invdeg(const int* __restrict__ deg,
                                                float* __restrict__ invdeg, int n) {
    int i = blockIdx.x * blockDim.x + threadIdx.x;
    if (i >= n) return;
    int d = deg[i];
    invdeg[i] = 1.0f / (float)(d > 1 ? d : 1);
}

// one-shot f32 -> bf16 conversion of all 9 weight matrices [W0|W1|W2][3][128][128]
__global__ __launch_bounds__(256) void k_cvt_w(const float* __restrict__ W0,
                                               const float* __restrict__ W1,
                                               const float* __restrict__ W2,
                                               __bf16* __restrict__ out) {
    const int PERW = 3 * DD * DD;                 // 49152 elems per weight tensor
    int t = blockIdx.x * blockDim.x + threadIdx.x;
    if (t >= 3 * PERW / 8) return;
    int i = t * 8;
    int which = i / PERW;
    int off = i - which * PERW;
    const float* src = (which == 0) ? W0 : (which == 1) ? W1 : W2;
    const float4 x = *(const float4*)(src + off);
    const float4 y = *(const float4*)(src + off + 4);
    v8bf o;
    o[0] = (__bf16)x.x; o[1] = (__bf16)x.y; o[2] = (__bf16)x.z; o[3] = (__bf16)x.w;
    o[4] = (__bf16)y.x; o[5] = (__bf16)y.y; o[6] = (__bf16)y.z; o[7] = (__bf16)y.w;
    *(v8bf*)(out + i) = o;
}

// S1[dst] += h[src]; S2[src] += h[dst]   (thread = edge x 4 features)
__global__ __launch_bounds__(256) void k_scatter(const int* __restrict__ src,
                                                 const int* __restrict__ dst,
                                                 const float* __restrict__ H,
                                                 float* __restrict__ S1,
                                                 float* __restrict__ S2, int E) {
    int t = blockIdx.x * blockDim.x + threadIdx.x;
    if (t >= E * 32) return;
    int e = t >> 5, d = (t & 31) * 4;
    int s = src[e], q = dst[e];
    const float4 hs = *(const float4*)(H + (size_t)s * DD + d);
    const float4 hd = *(const float4*)(H + (size_t)q * DD + d);
    float* p1 = S1 + (size_t)q * DD + d;
    float* p2 = S2 + (size_t)s * DD + d;
    atomicAdd(p1 + 0, hs.x); atomicAdd(p1 + 1, hs.y);
    atomicAdd(p1 + 2, hs.z); atomicAdd(p1 + 3, hs.w);
    atomicAdd(p2 + 0, hd.x); atomicAdd(p2 + 1, hd.y);
    atomicAdd(p2 + 2, hd.z); atomicAdd(p2 + 3, hd.w);
}

// ---------------------------------------------------------------------------
// Hout = relu(H@W0^T + (S1@W1^T + S2@W2^T) * invdeg[row])
// Block = 256 threads = 8 waves. Block owns a 16-row stripe: it converts the
// H/S1/S2 rows to bf16 in LDS once; wave w computes output cols [16w,16w+16)
// with 12 WMMAs (K=128 -> 4 steps x 3 matrices). Weights come pre-converted.
__global__ __launch_bounds__(256) void k_layer(const float* __restrict__ Hin,
                                               const float* __restrict__ S1,
                                               const float* __restrict__ S2,
                                               const __bf16* __restrict__ Wb0,
                                               const __bf16* __restrict__ Wb1,
                                               const __bf16* __restrict__ Wb2,
                                               const float* __restrict__ invdeg,
                                               float* __restrict__ Hout) {
    __shared__ __attribute__((aligned(16))) __bf16 sA[3][16][APAD];

    const int tid  = threadIdx.x;
    const int tm   = blockIdx.x;            // tile_m
    const int wave = tid >> 5;              // tile_n: 0..7
    const int lane = tid & 31;
    const int rsel = lane & 15;
    const int khiA = (lane & 16) ? 16 : 0;  // A-frag K offset per lane half
    const int khiB = (lane & 16);           // B-frag K offset per lane half

    // --- cooperative stage: 3 matrices x 16 rows x 128 cols, 8 elems/thread ---
    {
        const int row = tid >> 4;           // 0..15
        const int c0  = (tid & 15) * 8;     // 0..120
        const size_t g = (size_t)(tm * 16 + row) * DD + c0;
        const float* gp[3] = { Hin + g, S1 + g, S2 + g };
#pragma unroll
        for (int m = 0; m < 3; ++m) {
            const float4 x = *(const float4*)(gp[m]);
            const float4 y = *(const float4*)(gp[m] + 4);
            v8bf o;
            o[0] = (__bf16)x.x; o[1] = (__bf16)x.y; o[2] = (__bf16)x.z; o[3] = (__bf16)x.w;
            o[4] = (__bf16)y.x; o[5] = (__bf16)y.y; o[6] = (__bf16)y.z; o[7] = (__bf16)y.w;
            *(v8bf*)&sA[m][row][c0] = o;
        }
    }
    __syncthreads();

    const __bf16* w0row = Wb0 + (size_t)(wave * 16 + rsel) * DD;
    const __bf16* w1row = Wb1 + (size_t)(wave * 16 + rsel) * DD;
    const __bf16* w2row = Wb2 + (size_t)(wave * 16 + rsel) * DD;

    v8f acc0 = {};  // H @ W0^T
    v8f acc1 = {};  // S1 @ W1^T + S2 @ W2^T
#pragma unroll
    for (int s = 0; s < 4; ++s) {
        // A fragments from LDS: two 16-byte chunks per lane
        const __bf16* pH  = &sA[0][rsel][32 * s + khiA];
        const __bf16* pS1 = &sA[1][rsel][32 * s + khiA];
        const __bf16* pS2 = &sA[2][rsel][32 * s + khiA];
        v8bf hl = *(const v8bf*)pH,  hh = *(const v8bf*)(pH + 8);
        v8bf a1l = *(const v8bf*)pS1, a1h = *(const v8bf*)(pS1 + 8);
        v8bf a2l = *(const v8bf*)pS2, a2h = *(const v8bf*)(pS2 + 8);
        v16bf aH = __builtin_shufflevector(hl, hh, 0,1,2,3,4,5,6,7,8,9,10,11,12,13,14,15);
        v16bf a1 = __builtin_shufflevector(a1l, a1h, 0,1,2,3,4,5,6,7,8,9,10,11,12,13,14,15);
        v16bf a2 = __builtin_shufflevector(a2l, a2h, 0,1,2,3,4,5,6,7,8,9,10,11,12,13,14,15);
        // B fragments: 32 contiguous bytes of pre-converted weights
        v16bf b0 = *(const v16bf*)(w0row + 32 * s + khiB);
        v16bf b1 = *(const v16bf*)(w1row + 32 * s + khiB);
        v16bf b2 = *(const v16bf*)(w2row + 32 * s + khiB);

        acc0 = __builtin_amdgcn_wmma_f32_16x16x32_bf16(false, aH, false, b0,
                                                       (short)0, acc0, false, false);
        acc1 = __builtin_amdgcn_wmma_f32_16x16x32_bf16(false, a1, false, b1,
                                                       (short)0, acc1, false, false);
        acc1 = __builtin_amdgcn_wmma_f32_16x16x32_bf16(false, a2, false, b2,
                                                       (short)0, acc1, false, false);
    }

    // C/D layout: VGPR r, lanes 0-15 -> M=r, lanes 16-31 -> M=r+8; N = lane&15
    const int col = wave * 16 + rsel;
    const int mhi = (lane & 16) ? 8 : 0;
#pragma unroll
    for (int r = 0; r < 8; ++r) {
        const int row = tm * 16 + r + mhi;
        float v = acc0[r] + acc1[r] * invdeg[row];
        Hout[(size_t)row * DD + col] = v > 0.0f ? v : 0.0f;
    }
}

// ---------------------------------------------------------------------------
// graph pooling
__global__ __launch_bounds__(256) void k_cnt(const int* __restrict__ idx,
                                             int* __restrict__ cnt, int n) {
    int i = blockIdx.x * blockDim.x + threadIdx.x;
    if (i < n) atomicAdd(&cnt[idx[i]], 1);
}

// sum + max per graph; h >= 0 after relu so uint-bit atomicMax is order-correct
__global__ __launch_bounds__(256) void k_pool(const int* __restrict__ idx,
                                              const float* __restrict__ H,
                                              float* __restrict__ Gsum,
                                              unsigned* __restrict__ Gmax, int n) {
    int t = blockIdx.x * blockDim.x + threadIdx.x;
    if (t >= n * 32) return;
    int i = t >> 5, d = (t & 31) * 4;
    int g = idx[i];
    const float4 v = *(const float4*)(H + (size_t)i * DD + d);
    float*    ps = Gsum + (size_t)g * DD + d;
    unsigned* pm = Gmax + (size_t)g * DD + d;
    atomicAdd(ps + 0, v.x); atomicAdd(ps + 1, v.y);
    atomicAdd(ps + 2, v.z); atomicAdd(ps + 3, v.w);
    atomicMax(pm + 0, __float_as_uint(v.x));
    atomicMax(pm + 1, __float_as_uint(v.y));
    atomicMax(pm + 2, __float_as_uint(v.z));
    atomicMax(pm + 3, __float_as_uint(v.w));
}

// out[g][o] = [mean(g), max(g)] @ Wout + bout   (f32, exact head)
__global__ __launch_bounds__(256) void k_head(const float* __restrict__ Gsum,
                                              const unsigned* __restrict__ Gmax,
                                              const int* __restrict__ cnt,
                                              const float* __restrict__ Wout,
                                              const float* __restrict__ bout,
                                              float* __restrict__ out) {
    int t = blockIdx.x * blockDim.x + threadIdx.x;
    if (t >= GG * OUTD) return;
    int g = t >> 6, o = t & 63;
    int c = cnt[g];
    float inv = 1.0f / (float)(c > 1 ? c : 1);
    float acc = bout[o];
#pragma unroll 4
    for (int d = 0; d < DD; ++d) {
        acc += Gsum[(size_t)g * DD + d] * inv * Wout[(size_t)d * OUTD + o];
        acc += __uint_as_float(Gmax[(size_t)g * DD + d]) * Wout[(size_t)(DD + d) * OUTD + o];
    }
    out[t] = acc;
}

// ---------------------------------------------------------------------------
extern "C" void kernel_launch(void* const* d_in, const int* in_sizes, int n_in,
                              void* d_out, int out_size, void* d_ws, size_t ws_size,
                              hipStream_t stream) {
    const int*   nodes = (const int*)d_in[0];
    const int*   edges = (const int*)d_in[1];
    const int*   gidx  = (const int*)d_in[2];
    const float* emb   = (const float*)d_in[4];
    const float* W0    = (const float*)d_in[5];
    const float* W1    = (const float*)d_in[6];
    const float* W2    = (const float*)d_in[7];
    const float* Wout  = (const float*)d_in[8];
    const float* bout  = (const float*)d_in[9];
    float* out = (float*)d_out;

    const int N = in_sizes[0];
    const int E = in_sizes[1] / 2;
    const int* src = edges;        // edges[0]
    const int* dst = edges + E;    // edges[1]

    // carve scratch from d_ws (256B aligned slabs)
    char* base = (char*)d_ws;
    size_t off = 0;
    auto take = [&](size_t bytes) -> void* {
        void* p = base + off;
        off += (bytes + 255) & ~(size_t)255;
        return p;
    };
    float*    H0     = (float*)take((size_t)N * DD * 4);
    float*    H1     = (float*)take((size_t)N * DD * 4);
    float*    S1     = (float*)take((size_t)N * DD * 4);
    float*    S2     = (float*)take((size_t)N * DD * 4);
    float*    invdeg = (float*)take((size_t)N * 4);
    int*      degi   = (int*)take((size_t)N * 4);
    float*    Gsum   = (float*)take((size_t)GG * DD * 4);
    unsigned* Gmax   = (unsigned*)take((size_t)GG * DD * 4);
    int*      cnt    = (int*)take((size_t)GG * 4);
    __bf16*   Wb     = (__bf16*)take((size_t)3 * 3 * DD * DD * 2);  // [W0|W1|W2][3][128][128]

    // zero all accumulators every call (graph-capture-safe memsets)
    (void)hipMemsetAsync(degi, 0, (size_t)N * 4, stream);
    (void)hipMemsetAsync(Gsum, 0, (size_t)GG * DD * 4, stream);
    (void)hipMemsetAsync(Gmax, 0, (size_t)GG * DD * 4, stream);
    (void)hipMemsetAsync(cnt,  0, (size_t)GG * 4, stream);

    const int PERW = 3 * DD * DD;  // 49152 elems per weight tensor
    k_cvt_w<<<(3 * PERW / 8 + 255) / 256, 256, 0, stream>>>(W0, W1, W2, Wb);
    k_gather<<<(N * 32 + 255) / 256, 256, 0, stream>>>(nodes, emb, H0, N);
    k_degree<<<(E + 255) / 256, 256, 0, stream>>>(src, dst, degi, E);
    k_invdeg<<<(N + 255) / 256, 256, 0, stream>>>(degi, invdeg, N);

    const float* Hin = H0;
    float* Hout = H1;
    for (int l = 0; l < 3; ++l) {
        (void)hipMemsetAsync(S1, 0, (size_t)N * DD * 4, stream);
        (void)hipMemsetAsync(S2, 0, (size_t)N * DD * 4, stream);
        k_scatter<<<(E * 32 + 255) / 256, 256, 0, stream>>>(src, dst, Hin, S1, S2, E);
        k_layer<<<N / 16, 256, 0, stream>>>(Hin, S1, S2,
                                            Wb + 0 * PERW + (size_t)l * DD * DD,
                                            Wb + 1 * PERW + (size_t)l * DD * DD,
                                            Wb + 2 * PERW + (size_t)l * DD * DD,
                                            invdeg, Hout);
        const float* tmp = Hin; Hin = Hout; Hout = (float*)tmp;
    }

    k_cnt<<<(N + 255) / 256, 256, 0, stream>>>(gidx, cnt, N);
    k_pool<<<(N * 32 + 255) / 256, 256, 0, stream>>>(gidx, Hin, Gsum, Gmax, N);
    k_head<<<(GG * OUTD + 255) / 256, 256, 0, stream>>>(Gsum, Gmax, cnt, Wout, bout, out);
}
